// Encoder_60112362275061
// MI455X (gfx1250) — matrix operand order repaired and checked by reading, not compiled, not run on hardware
//
#include <hip/hip_runtime.h>
#include <hip/hip_bf16.h>

// Problem sizes (fixed by the reference)
#define Bn   128
#define Tn   1024
#define Xd   256
#define Hd   512

#define NWG  32                 // one workgroup per 16 hidden columns
#define WIH_STRIDE 272          // 256 + 16 pad bf16 elems -> 544B row (32B multiple)
#define WHH_STRIDE 528          // 512 + 16 pad bf16 elems -> 1056B row (32B multiple)
#define LDS_WIH_ELEMS (48 * WIH_STRIDE)
#define LDS_WHH_ELEMS (48 * WHH_STRIDE)
#define LDS_BYTES ((LDS_WIH_ELEMS + LDS_WHH_ELEMS) * 2)   // 76800 B < 320KB WGP LDS

typedef __attribute__((ext_vector_type(16))) __bf16 v16bf;
typedef __attribute__((ext_vector_type(8)))  __bf16 v8bf;
typedef __attribute__((ext_vector_type(8)))  float  v8f;

// Workspace layout (bytes)
#define WS_WIH   0                       // 1536*256*2 = 786432
#define WS_WHH   786432                  // 1536*512*2 = 1572864
#define WS_HF    2359296                 // 2*128*512*4 = 524288 (double-buffered f32 h)
#define WS_HB    2883584                 // 2*128*512*2 = 262144 (double-buffered bf16 h)
#define WS_BAR   3145728                 // barrier counter

static __device__ __forceinline__ v8f wmma_bf16(v16bf a, v16bf b, v8f c) {
    // D = A(16x32 bf16) x B(32x16 bf16) + C(16x16 f32)
    return __builtin_amdgcn_wmma_f32_16x16x32_bf16(
        /*neg_a=*/false, a, /*neg_b=*/false, b,
        /*c_mod=*/(short)0, c, /*reuse_a=*/false, /*reuse_b=*/false);
}

static __device__ __forceinline__ float sigmoid_f(float x) {
    return 1.0f / (1.0f + __expf(-x));
}

// ---------------------------------------------------------------------------
// Phase 0: convert weights to bf16, zero initial hidden state, reset barrier.
// Re-run every launch so graph replays are deterministic.
// ---------------------------------------------------------------------------
__global__ void gru_prep_kernel(const float* __restrict__ W_ih,
                                const float* __restrict__ W_hh,
                                __bf16* __restrict__ wih_b,
                                __bf16* __restrict__ whh_b,
                                float* __restrict__ hf,
                                __bf16* __restrict__ hb,
                                unsigned* __restrict__ bar) {
    const int tid = blockIdx.x * blockDim.x + threadIdx.x;
    const int stride = gridDim.x * blockDim.x;
    for (int i = tid; i < 3 * Hd * Xd; i += stride) wih_b[i] = (__bf16)W_ih[i];
    for (int i = tid; i < 3 * Hd * Hd; i += stride) whh_b[i] = (__bf16)W_hh[i];
    for (int i = tid; i < Bn * Hd; i += stride) {      // zero buffer 0 only (t=0 reads it)
        hf[i] = 0.0f;
        hb[i] = (__bf16)0.0f;
    }
    if (tid == 0) *bar = 0u;
}

// ---------------------------------------------------------------------------
// Phase 1: persistent fused GRU scan.
//   grid = 32 WGs x 256 threads (8 wave32). WG b owns hidden cols [16b,16b+16).
//   Wave w owns batch rows [16w,16w+16). Weights LDS-resident for all 1024
//   steps; h double-buffered in global (L2) with one grid barrier per step.
// ---------------------------------------------------------------------------
__global__ void __launch_bounds__(256, 1)
gru_scan_kernel(const float* __restrict__ x,        // [B,T,X]
                const float* __restrict__ b_ih,     // [3H]
                const float* __restrict__ b_hh,     // [3H]
                const __bf16* __restrict__ wih_b,   // [3H,X]
                const __bf16* __restrict__ whh_b,   // [3H,H]
                float* __restrict__ hf,             // [2,B,H] f32 state
                __bf16* __restrict__ hb,            // [2,B,H] bf16 state
                unsigned* __restrict__ bar,
                float* __restrict__ out) {          // [B,H]
    extern __shared__ __bf16 smem[];
    __bf16* lds_wih = smem;                   // [48][WIH_STRIDE]
    __bf16* lds_whh = smem + LDS_WIH_ELEMS;   // [48][WHH_STRIDE]

    const int tid  = threadIdx.x;
    const int w    = tid >> 5;        // wave id = M-tile (batch rows 16w..16w+15)
    const int lane = tid & 31;
    const int half = lane >> 4;       // K-half for A/B fragments
    const int nl   = lane & 15;       // A row-in-tile / B,N column-in-tile
    const int j0   = blockIdx.x * 16; // first hidden column owned by this WG
    const int j    = j0 + nl;         // this lane's output column

    // --- Stage weight slices into LDS (rows: gate g*16 + local col c) ---
    for (int i = tid; i < 48 * Xd; i += 256) {
        int rr = i / Xd, cc = i - rr * Xd;
        int grow = (rr >> 4) * Hd + j0 + (rr & 15);
        lds_wih[rr * WIH_STRIDE + cc] = wih_b[grow * Xd + cc];
    }
    for (int i = tid; i < 48 * Hd; i += 256) {
        int rr = i / Hd, cc = i - rr * Hd;
        int grow = (rr >> 4) * Hd + j0 + (rr & 15);
        lds_whh[rr * WHH_STRIDE + cc] = whh_b[grow * Hd + cc];
    }

    // Per-lane gate biases (column-constant)
    const float bir = b_ih[j],          bhr = b_hh[j];
    const float biz = b_ih[Hd + j],     bhz = b_hh[Hd + j];
    const float bin = b_ih[2 * Hd + j], bhn = b_hh[2 * Hd + j];

    const int m = 16 * w + nl;        // batch row this lane supplies for A
    __syncthreads();

    for (int t = 0; t < Tn; ++t) {
        const int cur = t & 1, nxt = cur ^ 1;
        v8f axr{}, axz{}, axn{};      // input-side gate accumulators
        v8f ahr{}, ahz{}, ahn{};      // hidden-side gate accumulators

        // ---- xg slice = x_t @ W_ih^T   (M=16, N=16x3, K=256) ----
        const float* xrow = x + ((size_t)m * Tn + t) * Xd;
        if (t + 1 < Tn) __builtin_prefetch(xrow + Xd, 0, 0);   // next step's x row
        #pragma unroll
        for (int k0 = 0; k0 < Xd; k0 += 32) {
            v8f f0 = *(const v8f*)(xrow + k0 + half * 16);
            v8f f1 = *(const v8f*)(xrow + k0 + half * 16 + 8);
            v16bf a;
            #pragma unroll
            for (int i = 0; i < 8; ++i) { a[i] = (__bf16)f0[i]; a[i + 8] = (__bf16)f1[i]; }
            const __bf16* wr = lds_wih + nl * WIH_STRIDE + k0 + half * 16;
            axr = wmma_bf16(a, *(const v16bf*)(wr), axr);
            axz = wmma_bf16(a, *(const v16bf*)(wr + 16 * WIH_STRIDE), axz);
            axn = wmma_bf16(a, *(const v16bf*)(wr + 32 * WIH_STRIDE), axn);
        }

        // ---- hg slice = h @ W_hh^T   (M=16, N=16x3, K=512) ----
        const __bf16* hrow = hb + (size_t)cur * Bn * Hd + (size_t)m * Hd;
        #pragma unroll
        for (int k0 = 0; k0 < Hd; k0 += 32) {
            v8bf lo = *(const v8bf*)(hrow + k0 + half * 16);
            v8bf hi = *(const v8bf*)(hrow + k0 + half * 16 + 8);
            v16bf a = __builtin_shufflevector(lo, hi,
                        0, 1, 2, 3, 4, 5, 6, 7, 8, 9, 10, 11, 12, 13, 14, 15);
            const __bf16* wr = lds_whh + nl * WHH_STRIDE + k0 + half * 16;
            ahr = wmma_bf16(a, *(const v16bf*)(wr), ahr);
            ahz = wmma_bf16(a, *(const v16bf*)(wr + 16 * WHH_STRIDE), ahz);
            ahn = wmma_bf16(a, *(const v16bf*)(wr + 32 * WHH_STRIDE), ahn);
        }

        // ---- fused gates + state update (C/D element r -> row 8*half + r) ----
        const float* hfp = hf + (size_t)cur * Bn * Hd;
        float*       hfn = hf + (size_t)nxt * Bn * Hd;
        __bf16*      hbn = hb + (size_t)nxt * Bn * Hd;
        #pragma unroll
        for (int r = 0; r < 8; ++r) {
            const int row = 16 * w + half * 8 + r;
            float rg = sigmoid_f(axr[r] + bir + ahr[r] + bhr);
            float zg = sigmoid_f(axz[r] + biz + ahz[r] + bhz);
            float ng = tanhf(axn[r] + bin + rg * (ahn[r] + bhn));
            float hp = hfp[(size_t)row * Hd + j];
            float hnew = (1.0f - zg) * ng + zg * hp;
            hfn[(size_t)row * Hd + j] = hnew;
            hbn[(size_t)row * Hd + j] = (__bf16)hnew;
            if (t == Tn - 1) out[(size_t)row * Hd + j] = hnew;
        }

        // ---- grid-wide barrier (release-arrive / acquire-spin, 1 per step) ----
        __threadfence();
        __syncthreads();
        if (tid == 0) {
            __hip_atomic_fetch_add(bar, 1u, __ATOMIC_ACQ_REL, __HIP_MEMORY_SCOPE_AGENT);
            const unsigned target = (unsigned)(t + 1) * NWG;
            while (__hip_atomic_load(bar, __ATOMIC_ACQUIRE, __HIP_MEMORY_SCOPE_AGENT) < target) {
                __builtin_amdgcn_s_sleep(4);
            }
        }
        __syncthreads();
        __threadfence();
    }
}

extern "C" void kernel_launch(void* const* d_in, const int* in_sizes, int n_in,
                              void* d_out, int out_size, void* d_ws, size_t ws_size,
                              hipStream_t stream) {
    (void)in_sizes; (void)n_in; (void)out_size; (void)ws_size;
    const float* x    = (const float*)d_in[0];   // [B,T,X]
    const float* W_ih = (const float*)d_in[1];   // [3H,X]
    const float* W_hh = (const float*)d_in[2];   // [3H,H]
    const float* b_ih = (const float*)d_in[3];   // [3H]
    const float* b_hh = (const float*)d_in[4];   // [3H]
    float* out = (float*)d_out;                  // [B,H]

    char* ws = (char*)d_ws;
    __bf16*   wih_b = (__bf16*)(ws + WS_WIH);
    __bf16*   whh_b = (__bf16*)(ws + WS_WHH);
    float*    hf    = (float*)(ws + WS_HF);
    __bf16*   hbuf  = (__bf16*)(ws + WS_HB);
    unsigned* bar   = (unsigned*)(ws + WS_BAR);

    gru_prep_kernel<<<768, 256, 0, stream>>>(W_ih, W_hh, wih_b, whh_b, hf, hbuf, bar);

    (void)hipFuncSetAttribute((const void*)gru_scan_kernel,
                              hipFuncAttributeMaxDynamicSharedMemorySize,
                              (int)LDS_BYTES);
    gru_scan_kernel<<<NWG, 256, LDS_BYTES, stream>>>(
        x, b_ih, b_hh, wih_b, whh_b, hf, hbuf, bar, out);
}
